// SelfAttention_65901978190161
// MI455X (gfx1250) — matrix operand order repaired and checked by reading, not compiled, not run on hardware
//
#include <hip/hip_runtime.h>
#include <hip/hip_bf16.h>

// Problem constants (from reference)
#define NB    2
#define TSEQ  4096
#define BT    8192            // NB*TSEQ
#define CDIM  768
#define NHEAD 12
#define HS    64
#define QKW   1536            // qk workspace row stride (q|k only)
#define VSTART 1536           // columns >= this in the 2304-wide GEMM go to vT

typedef __attribute__((ext_vector_type(16))) __bf16 v16bf;
typedef __attribute__((ext_vector_type(8)))  __bf16 v8bf;
typedef __attribute__((ext_vector_type(8)))  float  v8f;

static __device__ inline v8bf ld8(const __bf16* p) { return *(const v8bf*)p; }
static __device__ inline v16bf cat8(v8bf a, v8bf b) {
  return __builtin_shufflevector(a, b, 0,1,2,3,4,5,6,7,8,9,10,11,12,13,14,15);
}

// CDNA5 async global->LDS copy (ASYNCcnt), 16 bytes per lane.
static __device__ inline void async_ld_b128(unsigned lds_off, unsigned long long gaddr) {
  asm volatile("global_load_async_to_lds_b128 %0, %1, off" :: "v"(lds_off), "v"(gaddr) : "memory");
}

// ---------------------------------------------------------------------------
// Tiled GEMM: C = A[M,K] * B[K,N].  Block: 256 threads (8 waves), tile
// 128x128, K-step 32, double-buffered LDS software pipeline.
// Wave grid 4(m) x 2(n): each wave owns 32 rows x 64 cols = 2x4 WMMA tiles.
// AT==__bf16: A staged via global_load_async_to_lds_b128 (pure copy).
// AT==float : A loaded to regs, converted to bf16, stored to LDS.
// If vTp != nullptr, output columns >= VSTART are written transposed into
// vTp laid out as [(b*NHEAD+h)*HS + hs][TSEQ].
// ---------------------------------------------------------------------------
template <typename AT, typename OT>
__global__ __launch_bounds__(256)
void gemm_wmma(const AT* __restrict__ A, const float* __restrict__ Bw,
               OT* __restrict__ Co, __bf16* __restrict__ vTp,
               int M, int N, int K, int ldC)
{
  constexpr bool kAsyncA = (sizeof(AT) == 2);

  __shared__ __bf16 lds_a[2][128 * 40];   // [row][k] pad 40
  __shared__ __bf16 lds_b[2][128 * 40];   // [n][k]   transposed for B-frags

  const int tid  = threadIdx.x;
  const int lane = tid & 31;
  const int wv   = tid >> 5;
  const int hi   = lane >> 4;
  const int l15  = lane & 15;
  const int wm   = wv & 3;
  const int wn   = wv >> 2;
  const int m0   = blockIdx.y * 128;
  const int n0   = blockIdx.x * 128;

  v8f acc[2][4];
  #pragma unroll
  for (int i = 0; i < 2; ++i)
    #pragma unroll
    for (int j = 0; j < 4; ++j)
      #pragma unroll
      for (int e = 0; e < 8; ++e) acc[i][j][e] = 0.0f;

  const int arow = tid >> 1;           // 0..127
  const int akb  = (tid & 1) * 16;     // 0 or 16
  const int bk   = tid >> 3;           // 0..31
  const int bnb  = (tid & 7) * 16;     // 0..112

  auto issueAsyncA = [&](int kt, int b2) {
    unsigned l0 = (unsigned)(uintptr_t)&lds_a[b2][arow * 40 + akb];
    unsigned long long g0 =
        (unsigned long long)(uintptr_t)(A + (size_t)(m0 + arow) * K + kt + akb);
    async_ld_b128(l0, g0);
    async_ld_b128(l0 + 16u, g0 + 16ull);
  };

  float ar[16];                        // A staging regs (float path)
  float br[16];                        // B staging regs

  // ---- prologue: tile 0 in flight ----
  if constexpr (kAsyncA) {
    issueAsyncA(0, 0);
  } else {
    const AT* ap = A + (size_t)(m0 + arow) * K + akb;
    #pragma unroll
    for (int u = 0; u < 16; ++u) ar[u] = (float)ap[u];
  }
  {
    const float* bp = Bw + (size_t)bk * N + n0 + bnb;
    #pragma unroll
    for (int u = 0; u < 16; ++u) br[u] = bp[u];
  }

  int buf = 0;
  for (int kt = 0; kt < K; kt += 32) {
    // commit staged regs for tile kt into LDS[buf]
    if constexpr (!kAsyncA) {
      __bf16 tmp[16];
      #pragma unroll
      for (int u = 0; u < 16; ++u) tmp[u] = (__bf16)ar[u];
      *(v8bf*)&lds_a[buf][arow * 40 + akb]     = *(v8bf*)&tmp[0];
      *(v8bf*)&lds_a[buf][arow * 40 + akb + 8] = *(v8bf*)&tmp[8];
    }
    #pragma unroll
    for (int u = 0; u < 16; ++u)
      lds_b[buf][(bnb + u) * 40 + bk] = (__bf16)br[u];

    // issue loads for tile kt+32 (overlaps with this tile's WMMAs)
    const bool nxt = (kt + 32) < K;
    if (nxt) {
      if constexpr (kAsyncA) {
        issueAsyncA(kt + 32, buf ^ 1);
      } else {
        const AT* ap = A + (size_t)(m0 + arow) * K + kt + 32 + akb;
        #pragma unroll
        for (int u = 0; u < 16; ++u) ar[u] = (float)ap[u];
      }
      const float* bp = Bw + (size_t)(kt + 32 + bk) * N + n0 + bnb;
      #pragma unroll
      for (int u = 0; u < 16; ++u) br[u] = bp[u];
    }
    if constexpr (kAsyncA) {
      // async loads complete in order: <=2 outstanding means tile kt landed
      if (nxt) asm volatile("s_wait_asynccnt 0x2" ::: "memory");
      else     asm volatile("s_wait_asynccnt 0x0" ::: "memory");
    }
    __syncthreads();

    // ---- fragments + 8 WMMAs per wave ----
    v16bf afr[2], bfr[4];
    #pragma unroll
    for (int i = 0; i < 2; ++i) {
      // A 16x32: lanes 0-15 hold K 0-15; lanes 16-31 hold K 16-31
      const __bf16* ab = &lds_a[buf][(wm * 32 + i * 16 + l15) * 40];
      afr[i] = cat8(ld8(ab + hi * 16), ld8(ab + hi * 16 + 8));
    }
    #pragma unroll
    for (int j = 0; j < 4; ++j) {
      // B 32x16: lane = column, lanes 0-15 K 0-15, lanes 16-31 K 16-31
      const __bf16* bb = &lds_b[buf][(wn * 64 + j * 16 + l15) * 40 + hi * 16];
      bfr[j] = cat8(ld8(bb), ld8(bb + 8));
    }
    #pragma unroll
    for (int i = 0; i < 2; ++i)
      #pragma unroll
      for (int j = 0; j < 4; ++j)
        acc[i][j] = __builtin_amdgcn_wmma_f32_16x16x32_bf16(
            false, afr[i], false, bfr[j], (short)0, acc[i][j], false, false);
    __syncthreads();
    buf ^= 1;
  }

  // ---- epilogue: C/D layout — reg e: row = e + hi*8, col = l15 ----
  #pragma unroll
  for (int i = 0; i < 2; ++i)
    #pragma unroll
    for (int j = 0; j < 4; ++j) {
      const int row0 = m0 + wm * 32 + i * 16 + hi * 8;
      const int col  = n0 + wn * 64 + j * 16 + l15;
      if (vTp && col >= VSTART) {
        // transposed V store: vT[(b*NHEAD+h)*HS + hs][t]
        const int hh = (col - VSTART) >> 6;
        const int hs = (col - VSTART) & 63;
        #pragma unroll
        for (int e = 0; e < 8; ++e) {
          const int row = row0 + e;
          const int bb2 = row >> 12;        // / TSEQ
          const int t   = row & (TSEQ - 1);
          vTp[((size_t)(bb2 * NHEAD + hh) * HS + hs) * TSEQ + t] =
              (__bf16)(float)acc[i][j][e];
        }
      } else {
        #pragma unroll
        for (int e = 0; e < 8; ++e)
          Co[(size_t)(row0 + e) * ldC + col] = (OT)(float)acc[i][j][e];
      }
    }
}

// ---------------------------------------------------------------------------
// Flash attention (causal), barrier-free: block = 128 threads (4 waves), each
// wave owns 16 query rows and loops only over its own causal key range.
// qk rows of 1536: [q(768) | k(768)], each 768 = [head][64].
// vT: [(b*NHEAD+h)*HS + hs][TSEQ]  -> V B-fragments load contiguous keys.
// ---------------------------------------------------------------------------
__global__ __launch_bounds__(128)
void flash_attn(const __bf16* __restrict__ qk, const __bf16* __restrict__ vT,
                __bf16* __restrict__ aout)
{
  __shared__ __bf16 lds_p[4 * 16 * 40];    // per-wave P: [16 rows][32 cols pad40]

  const int tid  = threadIdx.x;
  const int lane = tid & 31;
  const int wv   = tid >> 5;
  const int hi   = lane >> 4;
  const int l15  = lane & 15;
  const int b    = blockIdx.z;
  const int h    = blockIdx.y;
  const int qrow0 = blockIdx.x * 64 + wv * 16;

  // Q fragments (16 rows x HS, two 32-wide K slices)
  v16bf qf[2];
  {
    const __bf16* qp = qk + (size_t)(b * TSEQ + qrow0 + l15) * QKW + h * HS;
    #pragma unroll
    for (int s = 0; s < 2; ++s)
      qf[s] = cat8(ld8(qp + 32 * s + hi * 16), ld8(qp + 32 * s + hi * 16 + 8));
  }
  const __bf16* vhbase = vT + (size_t)(b * NHEAD + h) * HS * TSEQ;
  const __bf16* khbase = qk + (size_t)(b * TSEQ) * QKW + CDIM + h * HS;

  v8f o[4];
  float mrow[8], lrow[8];
  #pragma unroll
  for (int t = 0; t < 4; ++t)
    #pragma unroll
    for (int e = 0; e < 8; ++e) o[t][e] = 0.0f;
  #pragma unroll
  for (int e = 0; e < 8; ++e) { mrow[e] = -1.0e30f; lrow[e] = 0.0f; }

  const int nch = (qrow0 + 47) >> 5;       // chunks covering keys 0..qrow0+15
  for (int c = 0; c < nch; ++c) {
    const int kc = c * 32;

    // prefetch next chunk's K rows (one per lane) and V^T rows (two per lane)
    if (c + 1 < nch) {
      __builtin_prefetch(khbase + (size_t)(kc + 32 + lane) * QKW, 0, 1);
      __builtin_prefetch(vhbase + (size_t)lane * TSEQ + kc + 32, 0, 1);
      __builtin_prefetch(vhbase + (size_t)(lane + 32) * TSEQ + kc + 32, 0, 1);
    }

    // ---- S = Q * K^T (two 16-key column tiles) ----
    v8f sa[2];
    #pragma unroll
    for (int j = 0; j < 2; ++j)
      #pragma unroll
      for (int e = 0; e < 8; ++e) sa[j][e] = 0.0f;
    #pragma unroll
    for (int j = 0; j < 2; ++j) {
      #pragma unroll
      for (int s = 0; s < 2; ++s) {
        // K fragment direct from global: a key's HS values are contiguous
        const __bf16* kp = khbase + (size_t)(kc + j * 16 + l15) * QKW
                           + 32 * s + hi * 16;
        v16bf kf = cat8(ld8(kp), ld8(kp + 8));
        sa[j] = __builtin_amdgcn_wmma_f32_16x16x32_bf16(
            false, qf[s], false, kf, (short)0, sa[j], false, false);
      }
    }
    const float scale = 0.125f;            // 1/sqrt(64)
    if (kc + 32 > qrow0) {                 // diagonal chunk: causal mask
      #pragma unroll
      for (int j = 0; j < 2; ++j) {
        const int key = kc + j * 16 + l15;
        #pragma unroll
        for (int e = 0; e < 8; ++e) {
          const int row = qrow0 + e + hi * 8;
          sa[j][e] = (key > row) ? -3.0e38f : sa[j][e] * scale;
        }
      }
    } else {
      #pragma unroll
      for (int j = 0; j < 2; ++j)
        #pragma unroll
        for (int e = 0; e < 8; ++e) sa[j][e] *= scale;
    }

    // ---- online softmax: reductions across 16 lanes sharing a row ----
    float mn[8], rs[8], al[8];
    #pragma unroll
    for (int e = 0; e < 8; ++e) {
      float cm = fmaxf(sa[0][e], sa[1][e]);
      cm = fmaxf(cm, __shfl_xor(cm, 1));
      cm = fmaxf(cm, __shfl_xor(cm, 2));
      cm = fmaxf(cm, __shfl_xor(cm, 4));
      cm = fmaxf(cm, __shfl_xor(cm, 8));
      mn[e] = fmaxf(mrow[e], cm);
      rs[e] = 0.0f;
    }
    __bf16* pbase = &lds_p[wv * 16 * 40];
    #pragma unroll
    for (int j = 0; j < 2; ++j)
      #pragma unroll
      for (int e = 0; e < 8; ++e) {
        float p = __expf(sa[j][e] - mn[e]);
        rs[e] += p;
        pbase[(e + hi * 8) * 40 + j * 16 + l15] = (__bf16)p;
      }
    #pragma unroll
    for (int e = 0; e < 8; ++e) {
      float r = rs[e];
      r += __shfl_xor(r, 1);
      r += __shfl_xor(r, 2);
      r += __shfl_xor(r, 4);
      r += __shfl_xor(r, 8);
      al[e] = __expf(mrow[e] - mn[e]);
      lrow[e] = lrow[e] * al[e] + r;
      mrow[e] = mn[e];
    }
    #pragma unroll
    for (int t = 0; t < 4; ++t)
      #pragma unroll
      for (int e = 0; e < 8; ++e) o[t][e] *= al[e];

    // wave-private P buffer: DS writes must retire before transposed re-read
    asm volatile("s_wait_dscnt 0" ::: "memory");

    // ---- O += P * V  (V fragments straight from vT, contiguous keys) ----
    const __bf16* pr = pbase + l15 * 40;
    v16bf pf = cat8(ld8(pr + hi * 16), ld8(pr + hi * 16 + 8));
    #pragma unroll
    for (int t = 0; t < 4; ++t) {
      const __bf16* vb = vhbase + (size_t)(t * 16 + l15) * TSEQ + kc + hi * 16;
      v16bf vf = cat8(ld8(vb), ld8(vb + 8));
      o[t] = __builtin_amdgcn_wmma_f32_16x16x32_bf16(
          false, pf, false, vf, (short)0, o[t], false, false);
    }
  }

  // ---- epilogue: normalize, store bf16 attention output [B*T, 768] ----
  #pragma unroll
  for (int t = 0; t < 4; ++t) {
    const int col = h * HS + t * 16 + l15;
    #pragma unroll
    for (int e = 0; e < 8; ++e) {
      const int row = qrow0 + e + hi * 8;
      aout[(size_t)(b * TSEQ + row) * CDIM + col] = (__bf16)(o[t][e] / lrow[e]);
    }
  }
}

// ---------------------------------------------------------------------------
extern "C" void kernel_launch(void* const* d_in, const int* in_sizes, int n_in,
                              void* d_out, int out_size, void* d_ws, size_t ws_size,
                              hipStream_t stream)
{
  const float* x      = (const float*)d_in[0];   // [2,4096,768]
  const float* w_attn = (const float*)d_in[1];   // [768,2304]
  const float* w_proj = (const float*)d_in[2];   // [768,768]
  float* out = (float*)d_out;                    // [2,4096,768]

  __bf16* qk   = (__bf16*)d_ws;                        // [8192, 1536] bf16 (q|k)
  __bf16* vT   = qk + (size_t)BT * QKW;                // [24*64, 4096] bf16 (V^T)
  __bf16* aout = vT + (size_t)BT * CDIM;               // [8192, 768]  bf16

  // 1) qkv = x @ w_attn (M=8192,N=2304,K=768); V columns land transposed in vT
  gemm_wmma<float, __bf16><<<dim3((3 * CDIM) / 128, BT / 128), 256, 0, stream>>>(
      x, w_attn, qk, vT, BT, 3 * CDIM, CDIM, QKW);

  // 2) fused causal flash attention (barrier-free waves)
  flash_attn<<<dim3(TSEQ / 64, NHEAD, NB), 128, 0, stream>>>(qk, vT, aout);

  // 3) out = attn_out @ w_proj (M=8192,N=768,K=768), A staged via async LDS DMA
  gemm_wmma<__bf16, float><<<dim3(CDIM / 128, BT / 128), 256, 0, stream>>>(
      aout, w_proj, out, nullptr, BT, CDIM, CDIM, CDIM);
}
